// TransA_47278999994720
// MI455X (gfx1250) — matrix operand order repaired and checked by reading, not compiled, not run on hardware
//
#include <hip/hip_runtime.h>
#include <math.h>

typedef _Float16 f16;
typedef __attribute__((ext_vector_type(16))) _Float16 v16h;
typedef __attribute__((ext_vector_type(8)))  _Float16 v8h;
typedef __attribute__((ext_vector_type(8)))  float    v8f;

#define NODE_SIZE 200000
#define LINK_SIZE 1000
#define DIM       128
#define BATCH     8192
#define MARGIN_F  1.0f

// ---- workspace layout (bytes) ----
#define ACC_OFF   0u                               // 4 floats: margin_sum, wr_sq, node_sq, link_sq
#define MASK_OFF  256u                             // LINK_SIZE ints (touched-row mask)
#define DELTA_OFF 4352u                            // DIM*DIM f32 (64 KB)
#define ERRP_OFF  69888u                           // BATCH*DIM f32 (4 MB)
#define ERRN_OFF  (ERRP_OFF + BATCH*DIM*4u)        // BATCH*DIM f32
#define ERRPT_OFF (ERRN_OFF + BATCH*DIM*4u)        // DIM*BATCH f16 (2 MB, transposed)
#define ERRNT_OFF (ERRPT_OFF + DIM*BATCH*2u)       // DIM*BATCH f16
// total ~12.65 MB

// ---------------------------------------------------------------------------
__global__ void k_zero(float* acc, int* mask) {
    int t = threadIdx.x;
    if (t < 4)         acc[t]  = 0.0f;
    if (t < LINK_SIZE) mask[t] = 0;
}

// ---------------------------------------------------------------------------
// err_p/err_n (f32 row-major) + f16 transposed copies + touched mask
__global__ void k_err(const int* __restrict__ sp, const int* __restrict__ tp,
                      const int* __restrict__ sn, const int* __restrict__ tn,
                      const int* __restrict__ r,
                      const float* __restrict__ node_emb,
                      const float* __restrict__ link_emb,
                      float* __restrict__ errp, float* __restrict__ errn,
                      f16* __restrict__ errpt, f16* __restrict__ errnt,
                      int* __restrict__ mask) {
    int b = blockIdx.x;
    int d = threadIdx.x;
    int isp = sp[b], itp = tp[b], isn = sn[b], itn = tn[b], ir = r[b];
    float re = link_emb[(size_t)ir * DIM + d];
    float ep = fabsf(node_emb[(size_t)isp * DIM + d] + re - node_emb[(size_t)itp * DIM + d]);
    float en = fabsf(node_emb[(size_t)isn * DIM + d] + re - node_emb[(size_t)itn * DIM + d]);
    errp[(size_t)b * DIM + d] = ep;
    errn[(size_t)b * DIM + d] = en;
    errpt[(size_t)d * BATCH + b] = (f16)ep;
    errnt[(size_t)d * BATCH + b] = (f16)en;
    if (d == 0) mask[ir] = 1;   // benign race, all write 1
}

// ---------------------------------------------------------------------------
// delta[128][128] = errn^T errn - errp^T errp, via v_wmma_f32_16x16x32_f16.
// One wave (32 threads) per 16x16 output tile; 8x8 = 64 tiles; K = 8192.
// Operand layouts per CDNA5 ISA 7.12.2 (A 16x32 f16, B 32x16 f16, C/D 16x16 f32).
__global__ void k_syrk_wmma(const f16* __restrict__ errpt,
                            const f16* __restrict__ errnt,
                            float* __restrict__ delta) {
    const int tile = blockIdx.x;
    const int m0 = (tile >> 3) * 16;
    const int n0 = (tile & 7) * 16;
    const int l  = threadIdx.x;
    const int hi = l >> 4;        // lane group
    const int mn = l & 15;        // row for A, col for B

    const f16* ap_row = errpt + (size_t)(m0 + mn) * BATCH;
    const f16* bp_row = errpt + (size_t)(n0 + mn) * BATCH;
    const f16* an_row = errnt + (size_t)(m0 + mn) * BATCH;
    const f16* bn_row = errnt + (size_t)(n0 + mn) * BATCH;

    v8f accp = {};
    v8f accn = {};

    for (int k0 = 0; k0 < BATCH; k0 += 32) {
        // A operand: lane-lo holds K={0..7,16..23}, lane-hi holds K={8..15,24..31}
        v8h ap0 = *(const v8h*)(ap_row + k0 + hi * 8);
        v8h ap1 = *(const v8h*)(ap_row + k0 + 16 + hi * 8);
        v8h an0 = *(const v8h*)(an_row + k0 + hi * 8);
        v8h an1 = *(const v8h*)(an_row + k0 + 16 + hi * 8);
        v16h aP, aN;
#pragma unroll
        for (int i = 0; i < 8; ++i) {
            aP[i] = ap0[i]; aP[i + 8] = ap1[i];
            aN[i] = an0[i]; aN[i + 8] = an1[i];
        }
        // B operand: lane-lo holds K={0..7,16..23}, lane-hi holds K={8..15,24..31}
        v8h bp0 = *(const v8h*)(bp_row + k0 + hi * 8);
        v8h bp1 = *(const v8h*)(bp_row + k0 + 16 + hi * 8);
        v8h bn0 = *(const v8h*)(bn_row + k0 + hi * 8);
        v8h bn1 = *(const v8h*)(bn_row + k0 + 16 + hi * 8);
        v16h bP = __builtin_shufflevector(bp0, bp1, 0,1,2,3,4,5,6,7,8,9,10,11,12,13,14,15);
        v16h bN = __builtin_shufflevector(bn0, bn1, 0,1,2,3,4,5,6,7,8,9,10,11,12,13,14,15);

        accp = __builtin_amdgcn_wmma_f32_16x16x32_f16(false, aP, false, bP,
                                                      (short)0, accp, false, false);
        accn = __builtin_amdgcn_wmma_f32_16x16x32_f16(false, aN, false, bN,
                                                      (short)0, accn, false, false);
    }

    // C/D layout: VGPR v, lanes<16 -> row v, lanes>=16 -> row v+8; col = lane&15
#pragma unroll
    for (int v = 0; v < 8; ++v) {
        int row = m0 + v + hi * 8;
        int col = n0 + mn;
        delta[row * DIM + col] = accn[v] - accp[v];
    }
}

// ---------------------------------------------------------------------------
__device__ __forceinline__ float block_reduce(float v, float* red) {
    int t = threadIdx.x;
    red[t] = v;
    __syncthreads();
    for (int s = blockDim.x >> 1; s > 0; s >>= 1) {
        if (t < s) red[t] += red[t + s];
        __syncthreads();
    }
    return red[0];
}

// ---------------------------------------------------------------------------
// Per-sample quadratic forms with fused relu(Wr0[r]+delta); row r is always a
// touched row. pos = ep^T W ep, neg = en^T W en; margin relu + atomic add.
// One block (256 thr) per sample; Wr0 rows stay L2-resident (<=1000 unique).
__global__ void k_quad(const int* __restrict__ r,
                       const float* __restrict__ Wr0,
                       const float* __restrict__ delta,
                       const float* __restrict__ errp,
                       const float* __restrict__ errn,
                       float* __restrict__ acc) {
    __shared__ float ep_s[DIM];
    __shared__ float en_s[DIM];
    __shared__ float red[256];

    int b = blockIdx.x;
    int t = threadIdx.x;
    if (t < DIM) ep_s[t] = errp[(size_t)b * DIM + t];
    else         en_s[t - DIM] = errn[(size_t)b * DIM + (t - DIM)];
    __syncthreads();

    int ir = r[b];
    const float* W0 = Wr0 + (size_t)ir * DIM * DIM;

    int i  = t >> 1;            // row 0..127
    int j0 = (t & 1) * 64;      // half-row
    float sp_ = 0.0f, sn_ = 0.0f;
#pragma unroll 8
    for (int j = 0; j < 64; ++j) {
        int jj = j0 + j;
        float w = fmaxf(W0[i * DIM + jj] + delta[i * DIM + jj], 0.0f);
        sp_ += w * ep_s[jj];
        sn_ += w * en_s[jj];
    }
    float pos_part = ep_s[i] * sp_;
    float neg_part = en_s[i] * sn_;

    float pos = block_reduce(pos_part, red);
    __syncthreads();
    float neg = block_reduce(neg_part, red);
    if (t == 0) atomicAdd(&acc[0], fmaxf(pos - neg + MARGIN_F, 0.0f));
}

// ---------------------------------------------------------------------------
// sum of squares of Wr where touched rows are relu(Wr0+delta), others Wr0
__global__ void k_wr_sq(const float* __restrict__ Wr0,
                        const float* __restrict__ delta,
                        const int* __restrict__ mask,
                        float* __restrict__ acc) {
    __shared__ float red[256];
    const long n = (long)LINK_SIZE * DIM * DIM;
    float local = 0.0f;
    for (long idx = (long)blockIdx.x * blockDim.x + threadIdx.x; idx < n;
         idx += (long)gridDim.x * blockDim.x) {
        int row = (int)(idx >> 14);          // 16384 = 2^14 elems per row
        int e   = (int)(idx & 16383);
        float w = Wr0[idx];
        if (mask[row]) w = fmaxf(w + delta[e], 0.0f);
        local += w * w;
    }
    float s = block_reduce(local, red);
    if (threadIdx.x == 0) atomicAdd(acc, s);
}

// ---------------------------------------------------------------------------
__global__ void k_sumsq(const float* __restrict__ x, long n, float* __restrict__ acc) {
    __shared__ float red[256];
    float local = 0.0f;
    for (long idx = (long)blockIdx.x * blockDim.x + threadIdx.x; idx < n;
         idx += (long)gridDim.x * blockDim.x) {
        float v = x[idx];
        local += v * v;
    }
    float s = block_reduce(local, red);
    if (threadIdx.x == 0) atomicAdd(acc, s);
}

// ---------------------------------------------------------------------------
__global__ void k_final(const float* __restrict__ acc, float* __restrict__ out) {
    out[0] = acc[0] / (float)BATCH
           + sqrtf(acc[1]) / (float)LINK_SIZE
           + sqrtf(acc[2]) / (float)NODE_SIZE
           + sqrtf(acc[3]) / (float)LINK_SIZE;
}

// ---------------------------------------------------------------------------
extern "C" void kernel_launch(void* const* d_in, const int* in_sizes, int n_in,
                              void* d_out, int out_size, void* d_ws, size_t ws_size,
                              hipStream_t stream) {
    const int*   sp       = (const int*)d_in[0];
    const int*   tp       = (const int*)d_in[1];
    const int*   sn       = (const int*)d_in[2];
    const int*   tn       = (const int*)d_in[3];
    const int*   r        = (const int*)d_in[4];
    const float* node_emb = (const float*)d_in[5];
    const float* link_emb = (const float*)d_in[6];
    const float* Wr0      = (const float*)d_in[7];
    float*       out      = (float*)d_out;

    unsigned char* ws = (unsigned char*)d_ws;
    float* acc   = (float*)(ws + ACC_OFF);
    int*   mask  = (int*)  (ws + MASK_OFF);
    float* delta = (float*)(ws + DELTA_OFF);
    float* errp  = (float*)(ws + ERRP_OFF);
    float* errn  = (float*)(ws + ERRN_OFF);
    f16*   errpt = (f16*)  (ws + ERRPT_OFF);
    f16*   errnt = (f16*)  (ws + ERRNT_OFF);

    k_zero<<<1, 1024, 0, stream>>>(acc, mask);

    k_err<<<BATCH, DIM, 0, stream>>>(sp, tp, sn, tn, r, node_emb, link_emb,
                                     errp, errn, errpt, errnt, mask);

    k_syrk_wmma<<<64, 32, 0, stream>>>(errpt, errnt, delta);

    k_quad<<<BATCH, 256, 0, stream>>>(r, Wr0, delta, errp, errn, acc);

    k_wr_sq<<<2048, 256, 0, stream>>>(Wr0, delta, mask, &acc[1]);

    k_sumsq<<<2048, 256, 0, stream>>>(node_emb, (long)NODE_SIZE * DIM, &acc[2]);
    k_sumsq<<<128,  256, 0, stream>>>(link_emb, (long)LINK_SIZE * DIM, &acc[3]);

    k_final<<<1, 1, 0, stream>>>(acc, out);
}
